// DemoTransformer_46222438040309
// MI455X (gfx1250) — matrix operand order repaired and checked by reading, not compiled, not run on hardware
//
#include <hip/hip_runtime.h>

// ---------------------------------------------------------------------------
// GPT-2 small forward on gfx1250 (MI455X): bf16 WMMA GEMMs + flash attention
// Weights pre-transposed to [N,K]; GEMM tiles double-buffered through LDS via
// async b128 global->LDS copies with threshold ASYNCcnt waits (SW pipeline).
// ---------------------------------------------------------------------------

typedef __attribute__((ext_vector_type(16))) __bf16          v16bf;
typedef __attribute__((ext_vector_type(16))) unsigned short  v16u;
typedef __attribute__((ext_vector_type(8)))  unsigned short  v8u;
typedef __attribute__((ext_vector_type(8)))  float           v8f;
typedef __attribute__((ext_vector_type(4)))  int             v4i;

__device__ __forceinline__ v16bf as_bf(v16u v) {
    return __builtin_bit_cast(v16bf, v);
}

__device__ __forceinline__ unsigned short f2bf_u(float f) {
    union { float f; unsigned u; } a; a.f = f;
    unsigned r = a.u + 0x7FFFu + ((a.u >> 16) & 1u);   // round-to-nearest-even
    return (unsigned short)(r >> 16);
}

__device__ __forceinline__ void storev(float* p, float v)          { *p = v; }
__device__ __forceinline__ void storev(unsigned short* p, float v) { *p = f2bf_u(v); }

__device__ __forceinline__ float gelu_new(float x) {
    float x3 = x * x * x;
    return 0.5f * x * (1.0f + tanhf(0.7978845608028654f * (x + 0.044715f * x3)));
}

__device__ __forceinline__ v8f wmma_bf16(v16u a, v16u b, v8f c) {
    return __builtin_amdgcn_wmma_f32_16x16x32_bf16(
        false, as_bf(a), false, as_bf(b), (short)0, c, false, false);
}

// 16-byte global -> LDS copy; async path if the toolchain exposes it.
__device__ __forceinline__ void async_copy_b128(const unsigned short* g,
                                                unsigned short* l) {
#if __has_builtin(__builtin_amdgcn_global_load_async_to_lds_b128)
    __builtin_amdgcn_global_load_async_to_lds_b128(
        (__attribute__((address_space(1))) v4i*)(size_t)g,
        (__attribute__((address_space(3))) v4i*)l,
        0, 0);
#else
    *(v8u*)l = *(const v8u*)g;
#endif
}

// Wait until at most N async copies from this wave are still outstanding.
template <int N>
__device__ __forceinline__ void wait_async_le() {
#if __has_builtin(__builtin_amdgcn_s_wait_asynccnt)
    __builtin_amdgcn_s_wait_asynccnt(N);
#else
    if (N == 0) asm volatile("s_wait_asynccnt 0" ::: "memory");
    else        asm volatile("s_wait_asynccnt 4" ::: "memory");
#endif
}

// ---------------------------------------------------------------------------
// Weight conversion / packing kernels (fp32 -> bf16, transposed to [N,K])
// ---------------------------------------------------------------------------

// src [L][K][N] f32 -> dst [L][Npad][K] bf16 (rows >= N zero-filled)
__global__ void convT_kernel(const float* __restrict__ src,
                             unsigned short* __restrict__ dst,
                             int Lc, int K, int N, int Npad) {
    long total = (long)Lc * Npad * K;
    long stride = (long)gridDim.x * blockDim.x;
    for (long i = (long)blockIdx.x * blockDim.x + threadIdx.x; i < total; i += stride) {
        int k = (int)(i % K);
        long rest = i / K;
        int n = (int)(rest % Npad);
        int l = (int)(rest / Npad);
        float v = 0.f;
        if (n < N) v = src[((size_t)l * K + k) * N + n];
        dst[i] = f2bf_u(v);
    }
}

// W_Q/K/V [L,H,D,DH] -> packed transposed [L][2304][768]; n = part*768+h*64+e
__global__ void pack_qkvT_kernel(const float* __restrict__ WQ,
                                 const float* __restrict__ WK,
                                 const float* __restrict__ WV,
                                 unsigned short* __restrict__ out, long n) {
    long stride = (long)gridDim.x * blockDim.x;
    for (long i = (long)blockIdx.x * blockDim.x + threadIdx.x; i < n; i += stride) {
        int d = (int)(i % 768);
        long rest = i / 768;
        int col = (int)(rest % 2304);
        int l = (int)(rest / 2304);
        int part = col / 768;
        int within = col % 768;
        int h = within >> 6, e = within & 63;
        const float* W = (part == 0) ? WQ : (part == 1) ? WK : WV;
        out[i] = f2bf_u(W[(((size_t)l * 12 + h) * 768 + d) * 64 + e]);
    }
}

__global__ void pack_qkv_bias_kernel(const float* __restrict__ bQ,
                                     const float* __restrict__ bK,
                                     const float* __restrict__ bV,
                                     float* __restrict__ out, int n) {
    int stride = gridDim.x * blockDim.x;
    for (int i = blockIdx.x * blockDim.x + threadIdx.x; i < n; i += stride) {
        int col = i % 2304, l = i / 2304;
        int part = col / 768, within = col % 768;
        const float* b = (part == 0) ? bQ : (part == 1) ? bK : bV;
        out[i] = b[l * 768 + within];
    }
}

// qkv [B*S, 2304] bf16 -> Qh,Kh [B*H, S, 64], Vt [B*H, 64, S]
__global__ void permute_qkv_kernel(const unsigned short* __restrict__ qkv,
                                   unsigned short* __restrict__ Qh,
                                   unsigned short* __restrict__ Kh,
                                   unsigned short* __restrict__ Vt) {
    long n = 2L * 1024 * 768;
    long stride = (long)gridDim.x * blockDim.x;
    for (long i = (long)blockIdx.x * blockDim.x + threadIdx.x; i < n; i += stride) {
        int within = (int)(i % 768);
        long bs = i / 768;
        int s = (int)(bs & 1023), b = (int)(bs >> 10);
        int h = within >> 6, e = within & 63;
        int bh = b * 12 + h;
        const unsigned short* src = qkv + (size_t)bs * 2304;
        Qh[((size_t)bh * 1024 + s) * 64 + e] = src[within];
        Kh[((size_t)bh * 1024 + s) * 64 + e] = src[768 + within];
        Vt[((size_t)bh * 64 + e) * 1024 + s] = src[1536 + within];
    }
}

// ---------------------------------------------------------------------------
// Embedding + LayerNorm
// ---------------------------------------------------------------------------

__global__ __launch_bounds__(256) void embed_kernel(const int* __restrict__ tokens,
                                                    const float* __restrict__ WE,
                                                    const float* __restrict__ Wpos,
                                                    float* __restrict__ resid) {
    int bs = blockIdx.x;
    int s = bs & 1023;
    int tok = tokens[bs];
    for (int i = threadIdx.x; i < 768; i += 256)
        resid[(size_t)bs * 768 + i] = WE[(size_t)tok * 768 + i] + Wpos[(size_t)s * 768 + i];
}

__global__ __launch_bounds__(256) void ln_kernel(const float* __restrict__ x,
                                                 const float* __restrict__ w,
                                                 const float* __restrict__ b,
                                                 unsigned short* __restrict__ out) {
    constexpr int D = 768;
    __shared__ float red[256];
    const int row = blockIdx.x, tid = threadIdx.x;
    const float* xr = x + (size_t)row * D;
    float s = 0.f;
    for (int i = tid; i < D; i += 256) s += xr[i];
    red[tid] = s; __syncthreads();
    for (int st = 128; st > 0; st >>= 1) { if (tid < st) red[tid] += red[tid + st]; __syncthreads(); }
    float mu = red[0] / D;
    __syncthreads();
    float v = 0.f;
    for (int i = tid; i < D; i += 256) { float d = xr[i] - mu; v += d * d; }
    red[tid] = v; __syncthreads();
    for (int st = 128; st > 0; st >>= 1) { if (tid < st) red[tid] += red[tid + st]; __syncthreads(); }
    float inv = rsqrtf(red[0] / D + 1e-5f);
    for (int i = tid; i < D; i += 256)
        out[(size_t)row * D + i] = f2bf_u((xr[i] - mu) * inv * w[i] + b[i]);
}

// ---------------------------------------------------------------------------
// bf16 WMMA GEMM: out[M,N] = A[M,K] * Wt[N,K]^T + bias (+res) (gelu?)
// 256 threads = 8 waves; block tile 128x128, K-step 32.
// Wave (wm = w&3, wn = w>>2) computes a 32x64 register tile: 8 WMMA / K-step.
// Double-buffered LDS; each thread issues 4 async b128 copies per tile and
// overlaps the next tile's copies with the current tile's WMMAs using a
// threshold s_wait_asynccnt (async loads retire in order).
// ---------------------------------------------------------------------------

template <int ACT, int ADD, typename OUT_T>
__global__ __launch_bounds__(256) void gemm_bf16_kernel(
    const unsigned short* __restrict__ A, const unsigned short* __restrict__ Wt,
    const float* __restrict__ bias, const float* __restrict__ res,
    OUT_T* __restrict__ out, int M, int N, int K) {
    __shared__ unsigned short Asub[2][128][40];
    __shared__ unsigned short Bsub[2][128][40];
    const int tid = threadIdx.x;
    const int wave = tid >> 5, lane = tid & 31;
    const int l15 = lane & 15, kh = lane >> 4;
    const int wm = wave & 3, wn = wave >> 2;
    const int m0 = blockIdx.y * 128, n0 = blockIdx.x * 128;
    (void)M;

    // this thread's two 16B chunks of each 128x32 tile (512 chunks total)
    const int chunk0 = tid * 2, chunk1 = tid * 2 + 1;
    const int row0 = chunk0 >> 2, q0 = (chunk0 & 3) * 8;
    const int row1 = chunk1 >> 2, q1 = (chunk1 & 3) * 8;

    auto issue = [&](int buf, int k0) {
        async_copy_b128(A  + (size_t)(m0 + row0) * K + k0 + q0, &Asub[buf][row0][q0]);
        async_copy_b128(A  + (size_t)(m0 + row1) * K + k0 + q1, &Asub[buf][row1][q1]);
        async_copy_b128(Wt + (size_t)(n0 + row0) * K + k0 + q0, &Bsub[buf][row0][q0]);
        async_copy_b128(Wt + (size_t)(n0 + row1) * K + k0 + q1, &Bsub[buf][row1][q1]);
    };

    v8f acc[2][4] = {};
    const int KT = K >> 5;
    issue(0, 0);
    for (int kt = 0; kt < KT; ++kt) {
        const int cur = kt & 1;
        if (kt + 1 < KT) {
            issue(cur ^ 1, (kt + 1) * 32);   // overlap next tile with this one
            wait_async_le<4>();              // in-order: retires current tile
        } else {
            wait_async_le<0>();
        }
        __syncthreads();

        v16u av[2];
        #pragma unroll
        for (int mi = 0; mi < 2; ++mi) {
            int arow = wm * 32 + mi * 16 + l15;
            #pragma unroll
            for (int j = 0; j < 8; ++j) {
                av[mi][j]     = Asub[cur][arow][kh * 16 + j];
                av[mi][8 + j] = Asub[cur][arow][kh * 16 + 8 + j];
            }
        }
        #pragma unroll
        for (int nt = 0; nt < 4; ++nt) {
            v16u bv;
            int brow = wn * 64 + nt * 16 + l15;
            #pragma unroll
            for (int j = 0; j < 16; ++j) bv[j] = Bsub[cur][brow][kh * 16 + j];
            acc[0][nt] = wmma_bf16(av[0], bv, acc[0][nt]);
            acc[1][nt] = wmma_bf16(av[1], bv, acc[1][nt]);
        }
        __syncthreads();   // all waves done reading buf[cur] before reuse
    }

    #pragma unroll
    for (int mi = 0; mi < 2; ++mi)
        #pragma unroll
        for (int nt = 0; nt < 4; ++nt)
            #pragma unroll
            for (int r = 0; r < 8; ++r) {
                int gM = m0 + wm * 32 + mi * 16 + r + 8 * kh;
                int gN = n0 + wn * 64 + nt * 16 + l15;
                if (gN < N) {
                    float v = acc[mi][nt][r] + bias[gN];
                    if (ACT) v = gelu_new(v);
                    if (ADD) v += res[(size_t)gM * N + gN];
                    storev(&out[(size_t)gM * N + gN], v);
                }
            }
}

// ---------------------------------------------------------------------------
// Flash attention: Q,K [B*H, S, 64] bf16, Vt [B*H, 64, S] bf16 -> Z [B,S,768]
// 128 threads = 4 waves, each wave owns a 16-query strip; k-tiles of 32.
// ---------------------------------------------------------------------------

__global__ __launch_bounds__(128) void attn_fa_kernel(
    const unsigned short* __restrict__ Q, const unsigned short* __restrict__ K,
    const unsigned short* __restrict__ Vt, unsigned short* __restrict__ Z) {
    constexpr int S = 1024, DH = 64, H = 12;
    __shared__ unsigned short Plds[4][16][32];
    const int wave = threadIdx.x >> 5, lane = threadIdx.x & 31;
    const int l15 = lane & 15, kh = lane >> 4;
    const int bh = blockIdx.y;                // b*H + h
    const int qrow0 = blockIdx.x * 64 + wave * 16;
    const unsigned short* Qb = Q  + (size_t)bh * S * DH;
    const unsigned short* Kb = K  + (size_t)bh * S * DH;
    const unsigned short* Vb = Vt + (size_t)bh * DH * S;

    // Q strip 16x64 as two 16x32 A fragments
    v16u qa[2];
    #pragma unroll
    for (int c = 0; c < 2; ++c) {
        const unsigned short* qp = Qb + (size_t)(qrow0 + l15) * DH + c * 32;
        #pragma unroll
        for (int j = 0; j < 8; ++j) {
            qa[c][j]     = qp[kh * 16 + j];
            qa[c][8 + j] = qp[kh * 16 + 8 + j];
        }
    }

    v8f o[4] = {};
    float mrow[8], lrow[8];
    #pragma unroll
    for (int r = 0; r < 8; ++r) { mrow[r] = -1e30f; lrow[r] = 0.f; }

    const int kend = qrow0 + 16;   // exclusive causal bound
    for (int kbase = 0; kbase < kend; kbase += 32) {
        v8f sc[2] = {};
        #pragma unroll
        for (int c = 0; c < 2; ++c) {      // feature-dim chunks of 32
            v16u kb0 = *(const v16u*)(Kb + (size_t)(kbase      + l15) * DH + c * 32 + kh * 16);
            v16u kb1 = *(const v16u*)(Kb + (size_t)(kbase + 16 + l15) * DH + c * 32 + kh * 16);
            sc[0] = wmma_bf16(qa[c], kb0, sc[0]);
            sc[1] = wmma_bf16(qa[c], kb1, sc[1]);
        }
        // scale + causal mask (C layout: M = r + 8*kh, N = l15 within tile)
        #pragma unroll
        for (int t = 0; t < 2; ++t) {
            int kpos = kbase + t * 16 + l15;
            #pragma unroll
            for (int r = 0; r < 8; ++r) {
                int srow = qrow0 + r + 8 * kh;
                float v = sc[t][r] * 0.125f;
                sc[t][r] = (kpos > srow) ? -1e5f : v;
            }
        }
        // online softmax: rows live across the 16-lane half
        float al[8];
        #pragma unroll
        for (int r = 0; r < 8; ++r) {
            float loc = fmaxf(sc[0][r], sc[1][r]);
            loc = fmaxf(loc, __shfl_xor(loc, 1));
            loc = fmaxf(loc, __shfl_xor(loc, 2));
            loc = fmaxf(loc, __shfl_xor(loc, 4));
            loc = fmaxf(loc, __shfl_xor(loc, 8));
            float mnew = fmaxf(mrow[r], loc);
            al[r] = __expf(mrow[r] - mnew);
            float e0 = __expf(sc[0][r] - mnew);
            float e1 = __expf(sc[1][r] - mnew);
            sc[0][r] = e0; sc[1][r] = e1;
            float ts = e0 + e1;
            ts += __shfl_xor(ts, 1);
            ts += __shfl_xor(ts, 2);
            ts += __shfl_xor(ts, 4);
            ts += __shfl_xor(ts, 8);
            lrow[r] = lrow[r] * al[r] + ts;
            mrow[r] = mnew;
        }
        #pragma unroll
        for (int t = 0; t < 4; ++t)
            #pragma unroll
            for (int r = 0; r < 8; ++r) o[t][r] *= al[r];

        // reshape P (C layout) -> A fragment through per-wave LDS
        __builtin_amdgcn_wave_barrier();
        #pragma unroll
        for (int t = 0; t < 2; ++t)
            #pragma unroll
            for (int r = 0; r < 8; ++r)
                Plds[wave][r + 8 * kh][t * 16 + l15] = f2bf_u(sc[t][r]);
        __builtin_amdgcn_wave_barrier();
        v16u pa;
        #pragma unroll
        for (int j = 0; j < 8; ++j) {
            pa[j]     = Plds[wave][l15][kh * 16 + j];
            pa[8 + j] = Plds[wave][l15][kh * 16 + 8 + j];
        }
        __builtin_amdgcn_wave_barrier();

        // O += P * V  (V pre-transposed so B fragments are contiguous)
        #pragma unroll
        for (int t = 0; t < 4; ++t) {
            v16u vb = *(const v16u*)(Vb + (size_t)(t * 16 + l15) * S + kbase + kh * 16);
            o[t] = wmma_bf16(pa, vb, o[t]);
        }
    }

    const int b = bh / H, h = bh % H;
    float inv[8];
    #pragma unroll
    for (int r = 0; r < 8; ++r) inv[r] = 1.0f / lrow[r];
    #pragma unroll
    for (int t = 0; t < 4; ++t)
        #pragma unroll
        for (int r = 0; r < 8; ++r) {
            int srow = qrow0 + r + 8 * kh;
            Z[(size_t)(b * S + srow) * (H * DH) + h * DH + t * 16 + l15] =
                f2bf_u(o[t][r] * inv[r]);
        }
}

// ---------------------------------------------------------------------------
// Host orchestration
// ---------------------------------------------------------------------------

extern "C" void kernel_launch(void* const* d_in, const int* in_sizes, int n_in,
                              void* d_out, int out_size, void* d_ws, size_t ws_size,
                              hipStream_t stream) {
    (void)in_sizes; (void)n_in; (void)out_size; (void)ws_size;
    constexpr int L = 12, H = 12, D = 768, DH = 64, DM = 3072, V = 50257;
    constexpr int Bsz = 2, S = 1024, M = Bsz * S;
    constexpr int Vpad = (V + 127) & ~127;       // 50304

    const int*   tokens = (const int*)  d_in[0];
    const float* W_E    = (const float*)d_in[1];
    const float* W_pos  = (const float*)d_in[2];
    const float* ln1_w  = (const float*)d_in[3];
    const float* ln1_b  = (const float*)d_in[4];
    const float* W_Q    = (const float*)d_in[5];
    const float* b_Q    = (const float*)d_in[6];
    const float* W_K    = (const float*)d_in[7];
    const float* b_K    = (const float*)d_in[8];
    const float* W_V    = (const float*)d_in[9];
    const float* b_V    = (const float*)d_in[10];
    const float* W_O    = (const float*)d_in[11];
    const float* b_O    = (const float*)d_in[12];
    const float* ln2_w  = (const float*)d_in[13];
    const float* ln2_b  = (const float*)d_in[14];
    const float* W_in   = (const float*)d_in[15];
    const float* b_in   = (const float*)d_in[16];
    const float* W_out  = (const float*)d_in[17];
    const float* b_out  = (const float*)d_in[18];
    const float* lnf_w  = (const float*)d_in[19];
    const float* lnf_b  = (const float*)d_in[20];
    const float* W_U    = (const float*)d_in[21];
    const float* b_U    = (const float*)d_in[22];

    char* base = (char*)d_ws;
    size_t off = 0;
    auto alloc = [&](size_t bytes) -> void* {
        void* p = base + off;
        off = (off + bytes + 255) & ~(size_t)255;
        return p;
    };
    unsigned short* wqkv = (unsigned short*)alloc((size_t)L * 2304 * D * 2);  // [L][2304][768]
    float*          bqkv = (float*)         alloc((size_t)L * 2304 * 4);
    unsigned short* wo   = (unsigned short*)alloc((size_t)L * D * D * 2);     // [L][768][768]
    unsigned short* win  = (unsigned short*)alloc((size_t)L * DM * D * 2);    // [L][3072][768]
    unsigned short* wout = (unsigned short*)alloc((size_t)L * D * DM * 2);    // [L][768][3072]
    unsigned short* wu   = (unsigned short*)alloc((size_t)Vpad * D * 2);      // [Vpad][768]
    float*          resid= (float*)         alloc((size_t)M * D * 4);
    unsigned short* xb   = (unsigned short*)alloc((size_t)M * D * 2);
    unsigned short* qkvt = (unsigned short*)alloc((size_t)M * 2304 * 2);
    unsigned short* Qh   = (unsigned short*)alloc((size_t)M * D * 2);
    unsigned short* Kh   = (unsigned short*)alloc((size_t)M * D * 2);
    unsigned short* Vt   = (unsigned short*)alloc((size_t)M * D * 2);
    unsigned short* z    = (unsigned short*)alloc((size_t)M * D * 2);
    unsigned short* hb   = (unsigned short*)alloc((size_t)M * DM * 2);

    auto cdiv = [](long a, long b) { return (int)((a + b - 1) / b); };

    // weight conversions -> bf16, transposed [N,K]
    pack_qkvT_kernel<<<cdiv((long)L * 2304 * D, 256), 256, 0, stream>>>(
        W_Q, W_K, W_V, wqkv, (long)L * 2304 * D);
    pack_qkv_bias_kernel<<<cdiv(L * 2304, 256), 256, 0, stream>>>(b_Q, b_K, b_V, bqkv, L * 2304);
    convT_kernel<<<cdiv((long)L * D * D, 256), 256, 0, stream>>>(W_O, wo, L, D, D, D);
    convT_kernel<<<cdiv((long)L * DM * D, 256), 256, 0, stream>>>(W_in, win, L, D, DM, DM);
    convT_kernel<<<cdiv((long)L * D * DM, 256), 256, 0, stream>>>(W_out, wout, L, DM, D, D);
    convT_kernel<<<cdiv((long)Vpad * D, 256), 256, 0, stream>>>(W_U, wu, 1, D, V, Vpad);

    embed_kernel<<<M, 256, 0, stream>>>(tokens, W_E, W_pos, resid);

    dim3 gQKV(2304 / 128, M / 128);
    dim3 gD(768 / 128, M / 128);
    dim3 gDM(3072 / 128, M / 128);
    for (int l = 0; l < L; ++l) {
        ln_kernel<<<M, 256, 0, stream>>>(resid, ln1_w + l * D, ln1_b + l * D, xb);
        gemm_bf16_kernel<0, 0, unsigned short><<<gQKV, 256, 0, stream>>>(
            xb, wqkv + (size_t)l * 2304 * D, bqkv + l * 2304, nullptr, qkvt, M, 2304, D);
        permute_qkv_kernel<<<cdiv((long)M * D, 256), 256, 0, stream>>>(qkvt, Qh, Kh, Vt);
        attn_fa_kernel<<<dim3(S / 64, Bsz * H), 128, 0, stream>>>(Qh, Kh, Vt, z);
        gemm_bf16_kernel<0, 1, float><<<gD, 256, 0, stream>>>(
            z, wo + (size_t)l * D * D, b_O + l * D, resid, resid, M, D, D);
        ln_kernel<<<M, 256, 0, stream>>>(resid, ln2_w + l * D, ln2_b + l * D, xb);
        gemm_bf16_kernel<1, 0, unsigned short><<<gDM, 256, 0, stream>>>(
            xb, win + (size_t)l * DM * D, b_in + l * DM, nullptr, hb, M, DM, D);
        gemm_bf16_kernel<0, 1, float><<<gD, 256, 0, stream>>>(
            hb, wout + (size_t)l * D * DM, b_out + l * D, resid, resid, M, D, DM);
    }
    ln_kernel<<<M, 256, 0, stream>>>(resid, lnf_w, lnf_b, xb);
    dim3 gV(Vpad / 128, M / 128);
    gemm_bf16_kernel<0, 0, float><<<gV, 256, 0, stream>>>(
        xb, wu, b_U, nullptr, (float*)d_out, M, V, D);
}
